// CrossVQEmbeddingEMA_60163901882670
// MI455X (gfx1250) — matrix operand order repaired and checked by reading, not compiled, not run on hardware
//
#include <hip/hip_runtime.h>
#include <hip/hip_bf16.h>
#include <stddef.h>

// ---------------------------------------------------------------------------
// Problem constants (from reference): B=64, T=512, D=256, M=1024
// ---------------------------------------------------------------------------
#define B_ 64
#define T_ 512
#define D_ 256
#define M_ 1024
#define N_ (B_ * T_)            // 32768
#define ND_ ((size_t)N_ * D_)   // 8388608

typedef __bf16 bf16_t;
typedef bf16_t v16bf __attribute__((ext_vector_type(16)));
typedef bf16_t v8bf  __attribute__((ext_vector_type(8)));
typedef float  v8f   __attribute__((ext_vector_type(8)));
typedef float  v4f   __attribute__((ext_vector_type(4)));

// ---------------- workspace layout (bytes) ----------------
#define EBF_OFF    ((size_t)0)          // bf16 embedding    512 KB
#define ENORM_OFF  ((size_t)524288)     // ||e||^2 per row     4 KB
#define APH_OFF    ((size_t)528384)     // a_ph [B,M] f32    256 KB   (zeroed)
#define EPH_OFF    ((size_t)790528)     // e_ph [B,M] f32    256 KB   (zeroed)
#define ACNT_OFF   ((size_t)1052672)    // a counts [B,M]    256 KB   (zeroed)
#define ECNT_OFF   ((size_t)1314816)    // e counts [B,M]    256 KB   (zeroed)
#define AHIST_OFF  ((size_t)1576960)    // a hist [M] int      4 KB   (zeroed)
#define EHIST_OFF  ((size_t)1581056)    // e hist [M] int      4 KB   (zeroed)
#define LACC_OFF   ((size_t)1585152)    // 4 loss accum f32           (zeroed)
#define EQ_OFF     ((size_t)1585168)    // equal_num int              (zeroed)
#define SCODE_OFF  ((size_t)1585184)    // Scode [64,64] f32  16 KB   (overwritten)
#define ACC_BYTES  (SCODE_OFF - APH_OFF)

// ---------------------------------------------------------------------------
// K1: embedding f32 -> bf16 + row norms.  One block per codebook row.
// ---------------------------------------------------------------------------
__global__ __launch_bounds__(256)
void prep_emb_kernel(const float* __restrict__ emb, bf16_t* __restrict__ ebf,
                     float* __restrict__ enorm2) {
  const int m = blockIdx.x;
  const int t = threadIdx.x;                 // 0..255 == d index
  float v = emb[(size_t)m * D_ + t];
  ebf[(size_t)m * D_ + t] = (bf16_t)v;
  float s = v * v;
#pragma unroll
  for (int msk = 1; msk < 32; msk <<= 1) s += __shfl_xor(s, msk, 32);
  __shared__ float wsum[8];
  if ((t & 31) == 0) wsum[t >> 5] = s;
  __syncthreads();
  if (t == 0) {
    float tot = 0.f;
#pragma unroll
    for (int i = 0; i < 8; ++i) tot += wsum[i];
    enorm2[m] = tot;
  }
}

// ---------------------------------------------------------------------------
// K2: fused VQ kernel.  One workgroup (256 thr = 8 waves) handles 16 rows of
// x against all M=1024 codes.  Each wave owns a 16x128 strip: 8 WMMA column
// tiles, K=256 in 8 steps of v_wmma_f32_16x16x32_bf16.  kt-outer / tile-inner
// with an explicit B double-buffer so each WMMA overlaps the next B load.
// The 16x1024 distance strip stays in VGPRs; softmax / argmin / ph fused.
// ---------------------------------------------------------------------------
__global__ __launch_bounds__(256)
void vq_main_kernel(const float* __restrict__ x,       // [N,D] f32
                    const float* __restrict__ embf32,  // [M,D] f32
                    const bf16_t* __restrict__ ebf,    // [M,D] bf16
                    const float* __restrict__ enorm2,  // [M]
                    float* __restrict__ qout,          // [N,D] quantized (ST fwd)
                    float* __restrict__ ph,            // [B,M] accum (pre-zeroed)
                    int* __restrict__ hist,            // [M]   accum
                    int* __restrict__ counts) {        // [B,M] accum
  __shared__ __align__(16) bf16_t aT[16][D_ + 8];   // staged A tile, bf16
  __shared__ float xnorm[16];
  __shared__ float wmin[8][16];
  __shared__ int   widx[8][16];
  __shared__ float wsum[8][16];
  __shared__ float rowmin[16];
  __shared__ int   rowidx[16];
  __shared__ float rowsum[16];

  const int t    = threadIdx.x;
  const int row0 = blockIdx.x * 16;
  const int b    = row0 / T_;

  // ---- stage A tile (f32 -> bf16) + per-row ||x||^2 (f32) ----
  {
    const int r  = t >> 4;             // 0..15
    const int c0 = (t & 15) * 16;
    const float* src = x + (size_t)(row0 + r) * D_ + c0;
    float s = 0.f;
#pragma unroll
    for (int i = 0; i < 16; ++i) {
      float v = src[i];
      s += v * v;
      aT[r][c0 + i] = (bf16_t)v;
    }
#pragma unroll
    for (int msk = 1; msk < 16; msk <<= 1) s += __shfl_xor(s, msk, 32);
    if ((t & 15) == 0) xnorm[r] = s;
  }
  __syncthreads();

  const int wv   = t >> 5;   // wave 0..7
  const int lane = t & 31;
  const int lh   = lane >> 4;  // half-wave
  const int ll   = lane & 15;
  const int nbase = wv * 128;

  // ---- K loop: 8 kt steps x 8 column tiles, B double-buffered ----
  v8f acc[8];
#pragma unroll
  for (int t8 = 0; t8 < 8; ++t8)
    acc[t8] = (v8f){0.f, 0.f, 0.f, 0.f, 0.f, 0.f, 0.f, 0.f};

  // B fragment base for this lane: column (nbase + t8*16 + ll), 16 contiguous K
  const bf16_t* bbase = ebf + (size_t)(nbase + ll) * D_ + lh * 16;
  const bf16_t* aptr  = &aT[ll][lh * 16];

#pragma unroll
  for (int kt = 0; kt < 8; ++kt) {
    // A fragment for this kt (shared by all 8 column tiles of this wave)
    v8bf a_lo = *(const v8bf*)(aptr + kt * 32);
    v8bf a_hi = *(const v8bf*)(aptr + kt * 32 + 8);
    v16bf af;
#pragma unroll
    for (int i = 0; i < 8; ++i) { af[i] = a_lo[i]; af[8 + i] = a_hi[i]; }

    // prime tile 0's B fragment
    v8bf bl = *(const v8bf*)(bbase + kt * 32);
    v8bf bh = *(const v8bf*)(bbase + kt * 32 + 8);
#pragma unroll
    for (int t8 = 0; t8 < 8; ++t8) {
      v8bf nbl, nbh;
      if (t8 < 7) {  // prefetch next tile's B while this tile's WMMA runs
        const bf16_t* nb = bbase + (size_t)(t8 + 1) * 16 * D_ + kt * 32;
        nbl = *(const v8bf*)(nb);
        nbh = *(const v8bf*)(nb + 8);
      }
      v16bf bv;
#pragma unroll
      for (int i = 0; i < 8; ++i) { bv[i] = bl[i]; bv[8 + i] = bh[i]; }
      acc[t8] = __builtin_amdgcn_wmma_f32_16x16x32_bf16(
          false, af, false, bv, (short)0, acc[t8], false, false);
      if (t8 < 7) { bl = nbl; bh = nbh; }
    }
  }

  // ---- epilogue: distances, per-row argmin candidates ----
  float dist[8][8];            // clamped distances, registers
  float bestd[8];              // raw min per row (argmin on unclamped)
  int   bestn[8];
#pragma unroll
  for (int i = 0; i < 8; ++i) { bestd[i] = 3.4e38f; bestn[i] = 0; }

#pragma unroll
  for (int t8 = 0; t8 < 8; ++t8) {
    const int n = nbase + t8 * 16 + ll;
    const float en = enorm2[n];
#pragma unroll
    for (int i = 0; i < 8; ++i) {
      const int m = i + lh * 8;
      float draw = xnorm[m] + en - 2.0f * acc[t8][i];
      dist[t8][i] = fmaxf(draw, 0.0f);
      if (draw < bestd[i] || (draw == bestd[i] && n < bestn[i])) {
        bestd[i] = draw; bestn[i] = n;
      }
    }
  }

  // ---- per-row argmin: 16-lane shuffle reduce, then cross-wave via LDS ----
#pragma unroll
  for (int i = 0; i < 8; ++i) {
    float d = bestd[i]; int n = bestn[i];
#pragma unroll
    for (int msk = 1; msk < 16; msk <<= 1) {
      float od = __shfl_xor(d, msk, 32);
      int   on = __shfl_xor(n, msk, 32);
      if (od < d || (od == d && on < n)) { d = od; n = on; }
    }
    if (ll == 0) { wmin[wv][i + lh * 8] = d; widx[wv][i + lh * 8] = n; }
  }
  __syncthreads();

  if (t < 16) {
    float d = wmin[0][t]; int n = widx[0][t];
#pragma unroll
    for (int w = 1; w < 8; ++w) {
      float od = wmin[w][t]; int on = widx[w][t];
      if (od < d || (od == d && on < n)) { d = od; n = on; }
    }
    rowmin[t] = fmaxf(d, 0.0f);
    rowidx[t] = n;
    atomicAdd(&hist[n], 1);
    atomicAdd(&counts[b * M_ + n], 1);
  }
  __syncthreads();

  // ---- gather quantized rows (ST forward value == embedding[argmin]) ----
  {
    const int r  = t >> 4;
    const int c0 = (t & 15) * 16;
    const float* er = embf32 + (size_t)rowidx[r] * D_ + c0;
    float* qo = qout + (size_t)(row0 + r) * D_ + c0;
#pragma unroll
    for (int i = 0; i < 16; i += 4) *(v4f*)(qo + i) = *(const v4f*)(er + i);
  }

  // ---- softmax denominator: rowmax(-sqrt d) = -sqrt(rowmin) ----
  float rmax[8];
#pragma unroll
  for (int i = 0; i < 8; ++i) rmax[i] = sqrtf(rowmin[i + lh * 8]);
  float se[8];
#pragma unroll
  for (int i = 0; i < 8; ++i) se[i] = 0.f;
#pragma unroll
  for (int t8 = 0; t8 < 8; ++t8)
#pragma unroll
    for (int i = 0; i < 8; ++i)
      se[i] += __expf(rmax[i] - sqrtf(dist[t8][i]));
#pragma unroll
  for (int i = 0; i < 8; ++i) {
    float s = se[i];
#pragma unroll
    for (int msk = 1; msk < 16; msk <<= 1) s += __shfl_xor(s, msk, 32);
    if (ll == 0) wsum[wv][i + lh * 8] = s;
  }
  __syncthreads();
  if (t < 16) {
    float s = 0.f;
#pragma unroll
    for (int w = 0; w < 8; ++w) s += wsum[w][t];
    rowsum[t] = s;
  }
  __syncthreads();

  // ---- ph accumulation: column-wise sum of softmax probs over 16 rows ----
  float rs[8];
#pragma unroll
  for (int i = 0; i < 8; ++i) rs[i] = rowsum[i + lh * 8];
#pragma unroll
  for (int t8 = 0; t8 < 8; ++t8) {
    float colp = 0.f;
#pragma unroll
    for (int i = 0; i < 8; ++i)
      colp += __expf(rmax[i] - sqrtf(dist[t8][i])) / rs[i];
    colp += __shfl_xor(colp, 16, 32);   // rows 8..15 live in partner half-wave
    if (lh == 0)
      atomicAdd(&ph[(size_t)b * M_ + nbase + t8 * 16 + ll], colp * (1.0f / T_));
  }
}

// ---------------------------------------------------------------------------
// K3: commitment-loss partial sums (4 squared-error accumulators).
// ---------------------------------------------------------------------------
__global__ __launch_bounds__(256)
void loss_kernel(const float* __restrict__ a, const float* __restrict__ e,
                 const float* __restrict__ qa, const float* __restrict__ qe,
                 float* __restrict__ acc) {
  const size_t i0 = ((size_t)blockIdx.x * 256 + threadIdx.x) * 8;
  float s0 = 0, s1 = 0, s2 = 0, s3 = 0;
#pragma unroll
  for (int k = 0; k < 8; ++k) {
    size_t i = i0 + k;
    float av = a[i], ev = e[i], qav = qa[i], qev = qe[i];
    float d;
    d = av - qav; s0 += d * d;
    d = av - qev; s1 += d * d;
    d = ev - qev; s2 += d * d;
    d = ev - qav; s3 += d * d;
  }
  float v[4] = {s0, s1, s2, s3};
  __shared__ float red[4][8];
#pragma unroll
  for (int q = 0; q < 4; ++q) {
    float s = v[q];
#pragma unroll
    for (int msk = 1; msk < 32; msk <<= 1) s += __shfl_xor(s, msk, 32);
    if ((threadIdx.x & 31) == 0) red[q][threadIdx.x >> 5] = s;
  }
  __syncthreads();
  if (threadIdx.x < 4) {
    float s = 0.f;
#pragma unroll
    for (int w = 0; w < 8; ++w) s += red[threadIdx.x][w];
    atomicAdd(&acc[threadIdx.x], s);
  }
}

// ---------------------------------------------------------------------------
// K4: per-sample mode of indices (argmax of counts, ties -> smallest index)
//     and equal_num accumulation.  One block per sample b.
// ---------------------------------------------------------------------------
__global__ __launch_bounds__(256)
void mode_kernel(const int* __restrict__ ca, const int* __restrict__ ce,
                 int* __restrict__ eqacc) {
  const int b = blockIdx.x, t = threadIdx.x;
  int bca = -1, bia = 0, bce = -1, bie = 0;
  for (int j = t; j < M_; j += 256) {
    int c = ca[b * M_ + j];
    if (c > bca) { bca = c; bia = j; }
    c = ce[b * M_ + j];
    if (c > bce) { bce = c; bie = j; }
  }
  // pack: larger count wins; tie -> smaller index wins
  __shared__ unsigned long long ra[256], re[256];
  ra[t] = ((unsigned long long)(unsigned)bca << 32) | (unsigned)(0xFFFFFFFFu - bia);
  re[t] = ((unsigned long long)(unsigned)bce << 32) | (unsigned)(0xFFFFFFFFu - bie);
  __syncthreads();
  for (int s = 128; s > 0; s >>= 1) {
    if (t < s) {
      if (ra[t + s] > ra[t]) ra[t] = ra[t + s];
      if (re[t + s] > re[t]) re[t] = re[t + s];
    }
    __syncthreads();
  }
  if (t == 0) {
    unsigned ia = 0xFFFFFFFFu - (unsigned)(ra[0] & 0xFFFFFFFFu);
    unsigned ie = 0xFFFFFFFFu - (unsigned)(re[0] & 0xFFFFFFFFu);
    if (ia == ie) atomicAdd(eqacc, 1);
  }
}

// ---------------------------------------------------------------------------
// K5: perplexities + commitment losses + equal_num -> scalar outputs [0..3],[5]
// ---------------------------------------------------------------------------
__global__ __launch_bounds__(256)
void finalize_kernel(const int* __restrict__ ha, const int* __restrict__ he,
                     const float* __restrict__ lacc, const int* __restrict__ eqacc,
                     float* __restrict__ outs) {
  const int t = threadIdx.x;
  float sa = 0.f, se = 0.f;
  for (int j = t; j < M_; j += 256) {
    float pa = (float)ha[j] * (1.0f / (float)N_);
    sa += pa * __logf(pa + 1e-10f);
    float pe = (float)he[j] * (1.0f / (float)N_);
    se += pe * __logf(pe + 1e-10f);
  }
#pragma unroll
  for (int msk = 1; msk < 32; msk <<= 1) {
    sa += __shfl_xor(sa, msk, 32);
    se += __shfl_xor(se, msk, 32);
  }
  __shared__ float rsa[8], rse[8];
  if ((t & 31) == 0) { rsa[t >> 5] = sa; rse[t >> 5] = se; }
  __syncthreads();
  if (t == 0) {
    float ta = 0.f, te = 0.f;
#pragma unroll
    for (int w = 0; w < 8; ++w) { ta += rsa[w]; te += rse[w]; }
    const float inv = 1.0f / (float)ND_;
    float Aaa = lacc[0] * inv, Aae = lacc[1] * inv;
    float Aee = lacc[2] * inv, Aea = lacc[3] * inv;
    outs[0] = 0.25f * (2.0f * Aaa + Aae);   // a_loss
    outs[1] = 0.25f * (2.0f * Aee + Aea);   // e_loss
    outs[2] = __expf(-ta);                  // a_perplexity
    outs[3] = __expf(-te);                  // e_perplexity
    outs[5] = (float)(*eqacc);              // equal_num
  }
}

// ---------------------------------------------------------------------------
// K6: Scode[b1][b2] = <a_ph[b1], log(e_ph[b2]+1e-10)> + <e_ph[b1], log(a_ph[b2]+1e-10)>
// ---------------------------------------------------------------------------
__global__ __launch_bounds__(128)
void scode_kernel(const float* __restrict__ aph, const float* __restrict__ eph,
                  float* __restrict__ sc) {
  const int b1 = blockIdx.x, b2 = blockIdx.y, t = threadIdx.x;
  float s = 0.f;
  for (int m = t; m < M_; m += 128)
    s += aph[b1 * M_ + m] * __logf(eph[b2 * M_ + m] + 1e-10f)
       + eph[b1 * M_ + m] * __logf(aph[b2 * M_ + m] + 1e-10f);
#pragma unroll
  for (int msk = 1; msk < 32; msk <<= 1) s += __shfl_xor(s, msk, 32);
  __shared__ float red[4];
  if ((t & 31) == 0) red[t >> 5] = s;
  __syncthreads();
  if (t == 0) sc[b1 * B_ + b2] = red[0] + red[1] + red[2] + red[3];
}

// ---------------------------------------------------------------------------
// K7: cmcm loss from Scode.  Single block.
// ---------------------------------------------------------------------------
__global__ __launch_bounds__(256)
void cmcm_kernel(const float* __restrict__ sc, float* __restrict__ out_cmcm) {
  const int t = threadIdx.x;
  float mn = 3.4e38f;
  for (int i = t; i < B_ * B_; i += 256) mn = fminf(mn, sc[i]);
#pragma unroll
  for (int msk = 1; msk < 32; msk <<= 1) mn = fminf(mn, __shfl_xor(mn, msk, 32));
  __shared__ float rmn[8];
  __shared__ float MaxS;
  if ((t & 31) == 0) rmn[t >> 5] = mn;
  __syncthreads();
  if (t == 0) {
    float m = rmn[0];
#pragma unroll
    for (int w = 1; w < 8; ++w) m = fminf(m, rmn[w]);
    MaxS = -m;                 // MaxScode = max(-Scode) = -min(Scode)
  }
  __syncthreads();
  __shared__ float lsum[64];
  if (t < B_) {
    float rs = 0.f, dg = 0.f;
    for (int j = 0; j < B_; ++j) {
      float v = __expf(sc[t * B_ + j] + MaxS);
      rs += v;
      if (j == t) dg = v;
    }
    lsum[t] = __logf(dg / (rs + 1e-5f));
  }
  __syncthreads();
  if (t == 0) {
    float tot = 0.f;
    for (int j = 0; j < B_; ++j) tot += lsum[j];
    *out_cmcm = 0.5f * (-(tot / (float)B_));   // 0.5 * Lcmcm
  }
}

// ---------------------------------------------------------------------------
extern "C" void kernel_launch(void* const* d_in, const int* in_sizes, int n_in,
                              void* d_out, int out_size, void* d_ws, size_t ws_size,
                              hipStream_t stream) {
  (void)in_sizes; (void)n_in; (void)out_size; (void)ws_size;
  const float* a   = (const float*)d_in[0];   // audio [B,T,D]
  const float* e   = (const float*)d_in[1];   // eeg   [B,T,D]
  const float* emb = (const float*)d_in[2];   // embedding [M,D]
  float* out = (float*)d_out;

  char* ws = (char*)d_ws;
  bf16_t* ebf   = (bf16_t*)(ws + EBF_OFF);
  float* enorm2 = (float*)(ws + ENORM_OFF);
  float* aph    = (float*)(ws + APH_OFF);
  float* eph    = (float*)(ws + EPH_OFF);
  int*   acnt   = (int*)(ws + ACNT_OFF);
  int*   ecnt   = (int*)(ws + ECNT_OFF);
  int*   ahist  = (int*)(ws + AHIST_OFF);
  int*   ehist  = (int*)(ws + EHIST_OFF);
  float* lacc   = (float*)(ws + LACC_OFF);
  int*   eqacc  = (int*)(ws + EQ_OFF);
  float* sc     = (float*)(ws + SCODE_OFF);

  float* qa = out;                 // a_quantized_st [N,D]
  float* qe = out + ND_;           // e_quantized_st [N,D]
  float* outs = out + 2 * ND_;     // [a_loss, e_loss, a_perp, e_perp, cmcm, equal]

  // zero all accumulators (harness does not re-poison between replays)
  hipMemsetAsync(ws + APH_OFF, 0, ACC_BYTES, stream);

  prep_emb_kernel<<<M_, 256, 0, stream>>>(emb, ebf, enorm2);

  vq_main_kernel<<<N_ / 16, 256, 0, stream>>>(a, emb, ebf, enorm2,
                                              qa, aph, ahist, acnt);
  vq_main_kernel<<<N_ / 16, 256, 0, stream>>>(e, emb, ebf, enorm2,
                                              qe, eph, ehist, ecnt);

  loss_kernel<<<(int)(ND_ / 2048), 256, 0, stream>>>(a, e, qa, qe, lacc);
  mode_kernel<<<B_, 256, 0, stream>>>(acnt, ecnt, eqacc);
  finalize_kernel<<<1, 256, 0, stream>>>(ahist, ehist, lacc, eqacc, outs);

  scode_kernel<<<dim3(B_, B_), 128, 0, stream>>>(aph, eph, sc);
  cmcm_kernel<<<1, 256, 0, stream>>>(sc, outs + 4);
}
